// Seq2SeqClassification_38671885534012
// MI455X (gfx1250) — compile-verified, matrix operand order checked by reading, not run on hardware
//
#include <hip/hip_runtime.h>

typedef __attribute__((ext_vector_type(16))) __bf16 v16bf;
typedef __attribute__((ext_vector_type(8)))  float  v8f;
typedef unsigned int   u32;
typedef unsigned short u16;
typedef __attribute__((ext_vector_type(4))) u32 u32x4;

namespace {

constexpr int   kB = 2, kS = 2048, kD = 2048;
constexpr int   kN  = kB * (kS - 1);   // 4094 shifted rows
constexpr int   kNP = 4096;            // row-padded (pad rows are zero)
constexpr float kEps    = 1e-5f;
constexpr float kNegBig = -1e30f;      // finite -inf sentinel (avoids NaN in merges)

// class counts: valid / padded-to-64
constexpr int HEAD_C = 1003, HEAD_CP = 1024;
constexpr int T1H = 512,  T1C = 1000,  T1CP = 1024;
constexpr int T2H = 128,  T2C = 3000,  T2CP = 3008;
constexpr int T3H = 32,   T3HP = 64,   T3C = 27000, T3CP = 27008;

constexpr int NT_H = HEAD_CP / 64, NT_1 = T1CP / 64, NT_2 = T2CP / 64, NT_3 = T3CP / 64;

// LDS tile strides (pad 32-elem rows to 40 elems = 80B to stagger banks)
constexpr int LDA = 40;
constexpr int A_TILE = 128 * LDA;   // elems per A buffer
constexpr int B_TILE = 64 * LDA;    // elems per B buffer

// ---------------- workspace layout (bytes, all chunks 256B-aligned) ----------------
constexpr size_t OFF_H    = 0;                               // h bf16 [4096 x 2048]
constexpr size_t OFF_HW   = OFF_H    + (size_t)kNP * kD * 2; // head_w bf16 [1024 x 2048]
constexpr size_t OFF_T1W1 = OFF_HW   + (size_t)HEAD_CP * kD * 2;
constexpr size_t OFF_T1W2 = OFF_T1W1 + (size_t)T1H * kD * 2;
constexpr size_t OFF_T2W1 = OFF_T1W2 + (size_t)T1CP * T1H * 2;
constexpr size_t OFF_T2W2 = OFF_T2W1 + (size_t)T2H * kD * 2;
constexpr size_t OFF_T3W1 = OFF_T2W2 + (size_t)T2CP * T2H * 2;
constexpr size_t OFF_T3W2 = OFF_T3W1 + (size_t)T3HP * kD * 2;
constexpr size_t OFF_T1HB = OFF_T3W2 + (size_t)T3CP * T3H * 2; // tail hiddens bf16
constexpr size_t OFF_T2HB = OFF_T1HB + (size_t)kNP * T1H * 2;
constexpr size_t OFF_T3HB = OFF_T2HB + (size_t)kNP * T2H * 2;
constexpr size_t OFF_PMH  = OFF_T3HB + (size_t)kNP * T3H * 2;  // lse partials (f32)
constexpr size_t OFF_PSH  = OFF_PMH  + (size_t)kNP * NT_H * 4;
constexpr size_t OFF_PM1  = OFF_PSH  + (size_t)kNP * NT_H * 4;
constexpr size_t OFF_PS1  = OFF_PM1  + (size_t)kNP * NT_1 * 4;
constexpr size_t OFF_PM2  = OFF_PS1  + (size_t)kNP * NT_1 * 4;
constexpr size_t OFF_PS2  = OFF_PM2  + (size_t)kNP * NT_2 * 4;
constexpr size_t OFF_PM3  = OFF_PS2  + (size_t)kNP * NT_2 * 4;
constexpr size_t OFF_PS3  = OFF_PM3  + (size_t)kNP * NT_3 * 4;
constexpr size_t OFF_ACC  = OFF_PS3  + (size_t)kNP * NT_3 * 4; // 1 float accumulator

} // namespace

// ---------------- bf16 helpers (round-to-nearest-even) ----------------
static __device__ __forceinline__ u16 f2bf(float f) {
  u32 x = __float_as_uint(f);
  x += ((x >> 16) & 1u) + 0x7FFFu;
  return (u16)(x >> 16);
}
static __device__ __forceinline__ float bf2f(u16 x) {
  return __uint_as_float(((u32)x) << 16);
}

// ---------------- gfx1250 async copy (global -> LDS), ASYNCcnt-tracked ----------------
static __device__ __forceinline__ void asyncLds16(const u16* gsrc, const u16* ldst) {
  const u32 ldsOff = (u32)(size_t)ldst;  // low 32 bits of a generic LDS pointer = LDS offset
  asm volatile("global_load_async_to_lds_b128 %0, %1, off"
               :: "v"(ldsOff), "v"(gsrc)
               : "memory");
}
static __device__ __forceinline__ void waitAsync0() {
  asm volatile("s_wait_asynccnt 0" ::: "memory");
}

// ---------------- wave32 reductions ----------------
static __device__ __forceinline__ float waveSum(float v) {
#pragma unroll
  for (int o = 16; o > 0; o >>= 1) v += __shfl_xor(v, o, 32);
  return v;
}
static __device__ __forceinline__ void lseMerge(float& M, float& S, float m, float s) {
  if (m > M) { S = S * __expf(M - m) + s; M = m; }
  else       { S += s * __expf(m - M); }
}
static __device__ __forceinline__ float waveLse(const float* pm, const float* ps,
                                                int n, int lane) {
  float M = kNegBig, S = 0.f;
  for (int t = lane; t < n; t += 32) lseMerge(M, S, pm[t], ps[t]);
#pragma unroll
  for (int o = 16; o > 0; o >>= 1) {
    float m2 = __shfl_xor(M, o, 32);
    float s2 = __shfl_xor(S, o, 32);
    lseMerge(M, S, m2, s2);
  }
  return M + __logf(S);
}

// bf16 dot product, one b128 per lane per step, wave-partial result
static __device__ __forceinline__ float dotBf(const u16* a, const u16* b, int n, int lane) {
  float s = 0.f;
  for (int k0 = lane * 8; k0 < n; k0 += 256) {
    u32x4 aq = *(const u32x4*)(a + k0);
    u32x4 bq = *(const u32x4*)(b + k0);
#pragma unroll
    for (int j = 0; j < 4; ++j)
      s += bf2f((u16)aq[j]) * bf2f((u16)bq[j]) +
           bf2f((u16)(aq[j] >> 16)) * bf2f((u16)(bq[j] >> 16));
  }
  return s;
}

// ---------------- WMMA fragment (CDNA5 16x16x32 bf16 layout) ----------------
// 16-bit A (16x32): lane L (m=L&15, g=L>>4) holds cols [g*8,g*8+8) and
// [16+g*8, 16+g*8+8) of row m -> two b128 reads. B (32x16 = 16 weight rows of
// length K) uses the identical pattern with the class row index in place of m.
union FragBf {
  v16bf v;
  u32x4 q[2];
};

// Per-wave core with LDS double-buffered async staging.
// Block tile: 128 rows x 64 cols; 8 waves; wave = 16 rows x 64 cols; K-step 32.
static __device__ __forceinline__ void wmmaTileLds(const u16* __restrict__ A,
                                                   const u16* __restrict__ W,
                                                   int K, int rowBase, int colBase,
                                                   u16* As, u16* Bs, v8f acc[4]) {
  const int tid  = threadIdx.x;
  const int wave = tid >> 5, lane = tid & 31;
  const int m = lane & 15, g = lane >> 4;
  const int nk = K >> 5;

  // stage one 128x32 A tile (2 chunks/thread) + one 64x32 B tile (1 chunk/thread)
  const int r0 = tid >> 2, cc0 = (tid & 3) * 8;          // chunk tid   -> rows 0..63
  const int c1 = tid + 256;
  const int r1 = c1 >> 2, cc1 = (c1 & 3) * 8;            // chunk tid+256 -> rows 64..127
  auto stage = [&](int k0, int buf) {
    u16* Ab = As + buf * A_TILE;
    u16* Bb = Bs + buf * B_TILE;
    asyncLds16(A + (size_t)(rowBase + r0) * K + k0 + cc0, Ab + r0 * LDA + cc0);
    asyncLds16(A + (size_t)(rowBase + r1) * K + k0 + cc1, Ab + r1 * LDA + cc1);
    asyncLds16(W + (size_t)(colBase + r0) * K + k0 + cc0, Bb + r0 * LDA + cc0);
  };

  stage(0, 0);
  for (int i = 0; i < nk; ++i) {
    const int buf = i & 1;
    waitAsync0();        // this wave's staged chunks for buf are in LDS
    __syncthreads();     // everyone's chunks are in LDS; prev-buf readers done
    if (i + 1 < nk) stage((i + 1) * 32, buf ^ 1);  // hide next copy behind WMMAs

    // Issue ALL fragment loads first so the LDS pipe runs ahead of the matrix
    // pipe (graduated s_wait_dscnt instead of dscnt==0 before every WMMA).
    const u16* ar = As + buf * A_TILE + (wave * 16 + m) * LDA;
    const u16* bb = Bs + buf * B_TILE + m * LDA;
    FragBf fa, fb0, fb1, fb2, fb3;
    fa.q[0]  = *(const u32x4*)(ar + g * 8);
    fa.q[1]  = *(const u32x4*)(ar + 16 + g * 8);
    fb0.q[0] = *(const u32x4*)(bb + 0 * 16 * LDA + g * 8);
    fb0.q[1] = *(const u32x4*)(bb + 0 * 16 * LDA + 16 + g * 8);
    fb1.q[0] = *(const u32x4*)(bb + 1 * 16 * LDA + g * 8);
    fb1.q[1] = *(const u32x4*)(bb + 1 * 16 * LDA + 16 + g * 8);
    fb2.q[0] = *(const u32x4*)(bb + 2 * 16 * LDA + g * 8);
    fb2.q[1] = *(const u32x4*)(bb + 2 * 16 * LDA + 16 + g * 8);
    fb3.q[0] = *(const u32x4*)(bb + 3 * 16 * LDA + g * 8);
    fb3.q[1] = *(const u32x4*)(bb + 3 * 16 * LDA + 16 + g * 8);
    acc[0] = __builtin_amdgcn_wmma_f32_16x16x32_bf16(
        false, fa.v, false, fb0.v, (short)0, acc[0], false, false);
    acc[1] = __builtin_amdgcn_wmma_f32_16x16x32_bf16(
        false, fa.v, false, fb1.v, (short)0, acc[1], false, false);
    acc[2] = __builtin_amdgcn_wmma_f32_16x16x32_bf16(
        false, fa.v, false, fb2.v, (short)0, acc[2], false, false);
    acc[3] = __builtin_amdgcn_wmma_f32_16x16x32_bf16(
        false, fa.v, false, fb3.v, (short)0, acc[3], false, false);
  }
  __syncthreads();
}

// ---------------- kernels ----------------

// LayerNorm (f32 in) -> bf16 shifted hidden rows; pad rows zeroed.
__global__ __launch_bounds__(256) void ln_kernel(const float* __restrict__ x,
                                                 const float* __restrict__ gamma,
                                                 const float* __restrict__ beta,
                                                 u16* __restrict__ h) {
  const int row = blockIdx.x;
  u16* hr = h + (size_t)row * kD;
  if (row >= kN) {
    for (int c = threadIdx.x; c < kD; c += 256) hr[c] = 0;
    return;
  }
  const int b = row / (kS - 1), s = row % (kS - 1);
  const float* xr = x + ((size_t)b * kS + s) * kD;
  __shared__ float r1v[256], r2v[256];
  float sum = 0.f, sq = 0.f;
  for (int c = threadIdx.x; c < kD; c += 256) {
    float v = xr[c];
    sum += v; sq += v * v;
  }
  r1v[threadIdx.x] = sum; r2v[threadIdx.x] = sq;
  __syncthreads();
  for (int o = 128; o > 0; o >>= 1) {
    if (threadIdx.x < o) { r1v[threadIdx.x] += r1v[threadIdx.x + o]; r2v[threadIdx.x] += r2v[threadIdx.x + o]; }
    __syncthreads();
  }
  const float mu  = r1v[0] * (1.f / kD);
  const float var = r2v[0] * (1.f / kD) - mu * mu;
  const float inv = rsqrtf(var + kEps);
  for (int c = threadIdx.x; c < kD; c += 256)
    hr[c] = f2bf((xr[c] - mu) * inv * gamma[c] + beta[c]);
}

// Cast f32 weights -> bf16, zero pad rows beyond rowsValid.
__global__ __launch_bounds__(256) void cast_pad_kernel(const float* __restrict__ src,
                                                       u16* __restrict__ dst,
                                                       int rowsValid, int total, int cols) {
  int i = blockIdx.x * 256 + threadIdx.x;
  if (i >= total) return;
  int r = i / cols;
  dst[i] = (r < rowsValid) ? f2bf(src[i]) : (u16)0;
}

// GEMM (bf16 WMMA, async-LDS staged) with fused online log-sum-exp epilogue.
__global__ __launch_bounds__(256) void gemm_lse_kernel(const u16* __restrict__ A,
                                                       const u16* __restrict__ W,
                                                       float* __restrict__ pMax,
                                                       float* __restrict__ pSum,
                                                       int K, int Cvalid, int nColTiles) {
  __shared__ u16 As[2 * A_TILE];
  __shared__ u16 Bs[2 * B_TILE];
  __shared__ float ldsC[128 * 64];
  const int wave = threadIdx.x >> 5, lane = threadIdx.x & 31;
  const int rowBase = blockIdx.y * 128, colBase = blockIdx.x * 64;
  v8f acc[4];
#pragma unroll
  for (int t = 0; t < 4; ++t) { v8f z = {}; acc[t] = z; }
  wmmaTileLds(A, W, K, rowBase, colBase, As, Bs, acc);
  const int g = lane >> 4, n = lane & 15;           // D-layout: lane holds col n, rows g*8..g*8+7
#pragma unroll
  for (int t = 0; t < 4; ++t)
#pragma unroll
    for (int r = 0; r < 8; ++r)
      ldsC[(wave * 16 + g * 8 + r) * 64 + t * 16 + n] = acc[t][r];
  __syncthreads();
  if (threadIdx.x < 128) {
    const int rl = threadIdx.x;
    float M = kNegBig, S = 0.f;
    for (int c = 0; c < 64; ++c) {
      if (colBase + c < Cvalid) lseMerge(M, S, ldsC[rl * 64 + c], 1.f);
    }
    const size_t row = (size_t)(rowBase + rl);
    pMax[row * nColTiles + blockIdx.x] = M;
    pSum[row * nColTiles + blockIdx.x] = S;
  }
}

// GEMM (bf16 WMMA, async-LDS staged) storing bf16 output (tail hidden projections).
__global__ __launch_bounds__(256) void gemm_store_kernel(const u16* __restrict__ A,
                                                         const u16* __restrict__ W,
                                                         u16* __restrict__ out,
                                                         int K, int Cvalid, int ldOut) {
  __shared__ u16 As[2 * A_TILE];
  __shared__ u16 Bs[2 * B_TILE];
  const int wave = threadIdx.x >> 5, lane = threadIdx.x & 31;
  const int rowBase = blockIdx.y * 128, colBase = blockIdx.x * 64;
  v8f acc[4];
#pragma unroll
  for (int t = 0; t < 4; ++t) { v8f z = {}; acc[t] = z; }
  wmmaTileLds(A, W, K, rowBase, colBase, As, Bs, acc);
  const int g = lane >> 4, n = lane & 15;
#pragma unroll
  for (int t = 0; t < 4; ++t) {
    const int col = colBase + t * 16 + n;
    if (col < Cvalid) {
#pragma unroll
      for (int r = 0; r < 8; ++r)
        out[(size_t)(rowBase + wave * 16 + g * 8 + r) * ldOut + col] = f2bf(acc[t][r]);
    }
  }
}

__global__ void zero_acc_kernel(float* acc) {
  if (threadIdx.x == 0 && blockIdx.x == 0) acc[0] = 0.f;
}

// One wave32 per row: merge LSE partials, recompute selected logits as dots,
// apply adaptive-softmax selection, atomic-add contribution.
__global__ __launch_bounds__(256) void finalize_kernel(
    const int* __restrict__ targets, const u16* __restrict__ h,
    const u16* __restrict__ headw,
    const u16* __restrict__ t1h, const u16* __restrict__ t1w2,
    const u16* __restrict__ t2h, const u16* __restrict__ t2w2,
    const u16* __restrict__ t3h, const u16* __restrict__ t3w2,
    const float* __restrict__ pmH, const float* __restrict__ psH,
    const float* __restrict__ pm1, const float* __restrict__ ps1,
    const float* __restrict__ pm2, const float* __restrict__ ps2,
    const float* __restrict__ pm3, const float* __restrict__ ps3,
    float* __restrict__ accOut) {
  const int wave = threadIdx.x >> 5, lane = threadIdx.x & 31;
  const int row = blockIdx.x * 8 + wave;
  if (row >= kN) return;
  const int b = row / (kS - 1), s = row % (kS - 1);
  const int lbl = targets[b * kS + s + 1];

  // log-sum-exps from per-tile partials
  const float lseH = waveLse(pmH + (size_t)row * NT_H, psH + (size_t)row * NT_H, NT_H, lane);
  const float lse1 = waveLse(pm1 + (size_t)row * NT_1, ps1 + (size_t)row * NT_1, NT_1, lane);
  const float lse2 = waveLse(pm2 + (size_t)row * NT_2, ps2 + (size_t)row * NT_2, NT_2, lane);
  const float lse3 = waveLse(pm3 + (size_t)row * NT_3, ps3 + (size_t)row * NT_3, NT_3, lane);

  // head dots: selected shortlist logit + 3 cluster logits (vectorized b128 reads)
  int csel = lbl < 1000 ? lbl : 999;
  if (csel < 0) csel = 0;
  const u16* hr  = h + (size_t)row * kD;
  const u16* w0p = headw + (size_t)csel * kD;
  const u16* w1p = headw + (size_t)1000 * kD;
  const u16* w2p = headw + (size_t)1001 * kD;
  const u16* w3p = headw + (size_t)1002 * kD;
  float d0 = 0.f, d1 = 0.f, d2 = 0.f, d3 = 0.f;
  for (int k0 = lane * 8; k0 < kD; k0 += 256) {
    u32x4 hq = *(const u32x4*)(hr + k0);
    u32x4 q0 = *(const u32x4*)(w0p + k0);
    u32x4 q1 = *(const u32x4*)(w1p + k0);
    u32x4 q2 = *(const u32x4*)(w2p + k0);
    u32x4 q3 = *(const u32x4*)(w3p + k0);
#pragma unroll
    for (int j = 0; j < 4; ++j) {
      const float hl = bf2f((u16)hq[j]), hh = bf2f((u16)(hq[j] >> 16));
      d0 += hl * bf2f((u16)q0[j]) + hh * bf2f((u16)(q0[j] >> 16));
      d1 += hl * bf2f((u16)q1[j]) + hh * bf2f((u16)(q1[j] >> 16));
      d2 += hl * bf2f((u16)q2[j]) + hh * bf2f((u16)(q2[j] >> 16));
      d3 += hl * bf2f((u16)q3[j]) + hh * bf2f((u16)(q3[j] >> 16));
    }
  }
  d0 = waveSum(d0); d1 = waveSum(d1); d2 = waveSum(d2); d3 = waveSum(d3);

  // tail selected logits
  int r1 = lbl - 1000; r1 = r1 < 0 ? 0 : (r1 > T1C - 1 ? T1C - 1 : r1);
  int r2 = lbl - 2000; r2 = r2 < 0 ? 0 : (r2 > T2C - 1 ? T2C - 1 : r2);
  int r3 = lbl - 5000; r3 = r3 < 0 ? 0 : (r3 > T3C - 1 ? T3C - 1 : r3);
  float td1 = waveSum(dotBf(t1h + (size_t)row * T1H, t1w2 + (size_t)r1 * T1H, T1H, lane));
  float td2 = waveSum(dotBf(t2h + (size_t)row * T2H, t2w2 + (size_t)r2 * T2H, T2H, lane));
  float td3 = waveSum(dotBf(t3h + (size_t)row * T3H, t3w2 + (size_t)r3 * T3H, T3H, lane));

  float outv;
  if (lbl < 1000)      outv = d0 - lseH;
  else if (lbl < 2000) outv = (d1 - lseH) + (td1 - lse1);
  else if (lbl < 5000) outv = (d2 - lseH) + (td2 - lse2);
  else                 outv = (d3 - lseH) + (td3 - lse3);

  if (lane == 0) atomicAdd(accOut, outv);
}

__global__ void finish_kernel(const float* __restrict__ acc, float* __restrict__ out) {
  if (threadIdx.x == 0 && blockIdx.x == 0) out[0] = -acc[0] / (float)kN;
}

// ---------------- host launch ----------------
extern "C" void kernel_launch(void* const* d_in, const int* in_sizes, int n_in,
                              void* d_out, int out_size, void* d_ws, size_t ws_size,
                              hipStream_t stream) {
  const float* x       = (const float*)d_in[0];
  const int*   targets = (const int*)d_in[1];
  const float* gamma   = (const float*)d_in[2];
  const float* beta    = (const float*)d_in[3];
  const float* head_w  = (const float*)d_in[4];
  const float* t1_w1   = (const float*)d_in[5];
  const float* t1_w2   = (const float*)d_in[6];
  const float* t2_w1   = (const float*)d_in[7];
  const float* t2_w2   = (const float*)d_in[8];
  const float* t3_w1   = (const float*)d_in[9];
  const float* t3_w2   = (const float*)d_in[10];

  char* ws = (char*)d_ws;
  u16* h      = (u16*)(ws + OFF_H);
  u16* hw     = (u16*)(ws + OFF_HW);
  u16* w11    = (u16*)(ws + OFF_T1W1);
  u16* w12    = (u16*)(ws + OFF_T1W2);
  u16* w21    = (u16*)(ws + OFF_T2W1);
  u16* w22    = (u16*)(ws + OFF_T2W2);
  u16* w31    = (u16*)(ws + OFF_T3W1);
  u16* w32    = (u16*)(ws + OFF_T3W2);
  u16* t1hB   = (u16*)(ws + OFF_T1HB);
  u16* t2hB   = (u16*)(ws + OFF_T2HB);
  u16* t3hB   = (u16*)(ws + OFF_T3HB);
  float* pmH  = (float*)(ws + OFF_PMH);
  float* psH  = (float*)(ws + OFF_PSH);
  float* pm1  = (float*)(ws + OFF_PM1);
  float* ps1  = (float*)(ws + OFF_PS1);
  float* pm2  = (float*)(ws + OFF_PM2);
  float* ps2  = (float*)(ws + OFF_PS2);
  float* pm3  = (float*)(ws + OFF_PM3);
  float* ps3  = (float*)(ws + OFF_PS3);
  float* acc  = (float*)(ws + OFF_ACC);

  auto blocksFor = [](int total) { return (total + 255) / 256; };

  // 0) zero loss accumulator
  zero_acc_kernel<<<1, 32, 0, stream>>>(acc);

  // 1) weight cast/pad to bf16
  cast_pad_kernel<<<blocksFor(HEAD_CP * kD), 256, 0, stream>>>(head_w, hw, HEAD_C, HEAD_CP * kD, kD);
  cast_pad_kernel<<<blocksFor(T1H * kD),     256, 0, stream>>>(t1_w1, w11, T1H, T1H * kD, kD);
  cast_pad_kernel<<<blocksFor(T1CP * T1H),   256, 0, stream>>>(t1_w2, w12, T1C, T1CP * T1H, T1H);
  cast_pad_kernel<<<blocksFor(T2H * kD),     256, 0, stream>>>(t2_w1, w21, T2H, T2H * kD, kD);
  cast_pad_kernel<<<blocksFor(T2CP * T2H),   256, 0, stream>>>(t2_w2, w22, T2C, T2CP * T2H, T2H);
  cast_pad_kernel<<<blocksFor(T3HP * kD),    256, 0, stream>>>(t3_w1, w31, T3H, T3HP * kD, kD);
  cast_pad_kernel<<<blocksFor(T3CP * T3H),   256, 0, stream>>>(t3_w2, w32, T3C, T3CP * T3H, T3H);

  // 2) LayerNorm -> bf16 hidden (with row padding)
  ln_kernel<<<kNP, 256, 0, stream>>>(x, gamma, beta, h);

  const int rowTiles = kNP / 128;  // 32

  // 3) head logits with fused LSE
  gemm_lse_kernel<<<dim3(NT_H, rowTiles), 256, 0, stream>>>(h, hw, pmH, psH, kD, HEAD_C, NT_H);

  // 4) tail hidden projections (store bf16)
  gemm_store_kernel<<<dim3(T1H / 64, rowTiles), 256, 0, stream>>>(h, w11, t1hB, kD, T1H, T1H);
  gemm_store_kernel<<<dim3(T2H / 64, rowTiles), 256, 0, stream>>>(h, w21, t2hB, kD, T2H, T2H);
  gemm_store_kernel<<<dim3(T3HP / 64, rowTiles), 256, 0, stream>>>(h, w31, t3hB, kD, T3H, T3H);

  // 5) tail class logits with fused LSE
  gemm_lse_kernel<<<dim3(NT_1, rowTiles), 256, 0, stream>>>(t1hB, w12, pm1, ps1, T1H, T1C, NT_1);
  gemm_lse_kernel<<<dim3(NT_2, rowTiles), 256, 0, stream>>>(t2hB, w22, pm2, ps2, T2H, T2C, NT_2);
  gemm_lse_kernel<<<dim3(NT_3, rowTiles), 256, 0, stream>>>(t3hB, w32, pm3, ps3, T3H, T3C, NT_3);

  // 6) per-row finalize (one wave32 per row) + scalar finish
  finalize_kernel<<<(kN + 7) / 8, 256, 0, stream>>>(
      targets, h, hw, t1hB, w12, t2hB, w22, t3hB, w32,
      pmH, psH, pm1, ps1, pm2, ps2, pm3, ps3, acc);
  finish_kernel<<<1, 32, 0, stream>>>(acc, (float*)d_out);
}
